// DROPEDGE_63763084476890
// MI455X (gfx1250) — matrix operand order repaired and checked by reading, not compiled, not run on hardware
//
#include <hip/hip_runtime.h>
#include <hip/hip_bf16.h>

// ---------------------------------------------------------------------------
// CDNA5 (gfx1250) GraphConv x3:  h' = relu( nd * SpMM( (h*ns) @ W ) + b )
// GEMM path uses v_wmma_f32_16x16x32_bf16 (wave32 WMMA); operands fetched
// from padded LDS as 16-byte ds_load_b128 pairs (no scalar u16 gathers).
// ---------------------------------------------------------------------------

typedef __attribute__((ext_vector_type(16))) __bf16 v16bf;
typedef __attribute__((ext_vector_type(8)))  __bf16 v8bf;
typedef __attribute__((ext_vector_type(8)))  float  v8f;

// ------------------------------- utility -----------------------------------

__global__ __launch_bounds__(256) void zero_f32_kernel(float* __restrict__ p, long long n) {
    long long i = (long long)blockIdx.x * blockDim.x + threadIdx.x;
    if (i < n) p[i] = 0.0f;
}

__global__ __launch_bounds__(256) void degree_kernel(const int* __restrict__ src,
                                                     const int* __restrict__ dst,
                                                     float* __restrict__ degO,
                                                     float* __restrict__ degI,
                                                     int E) {
    int e = blockIdx.x * blockDim.x + threadIdx.x;
    if (e >= E) return;
    atomicAdd(&degO[src[e]], 1.0f);
    atomicAdd(&degI[dst[e]], 1.0f);
}

// in-place: deg -> rsqrt(max(deg,1))   (applied to ns and nd arrays at once)
__global__ __launch_bounds__(256) void norm_kernel(float* __restrict__ deg, int n) {
    int i = blockIdx.x * blockDim.x + threadIdx.x;
    if (i >= n) return;
    deg[i] = rsqrtf(fmaxf(deg[i], 1.0f));
}

// ------------------------- WMMA GEMM with fused ns -------------------------
// out[N x KOUT] = (H[N x 128] * ns[N]) @ W[128 x KOUT]
// 128 threads (4 waves), 64 rows per block. LDS bf16 staging:
//   sA  : A tile row-major, padded stride (ns fused, zero-padded past N)
//   sWt : W transposed ([j][k]) so B operands are contiguous 32B runs.
template <int KOUT>
__global__ __launch_bounds__(128) void gemm_ns_kernel(const float* __restrict__ H,
                                                      const float* __restrict__ W,
                                                      const float* __restrict__ ns,
                                                      float* __restrict__ out,
                                                      int N) {
    constexpr int LDP = 136;            // padded LDS stride in bf16 (272B: keeps
                                        // 16B alignment, staggers banks by 4)
    __shared__ __bf16 sA [64   * LDP];  // 17,408 B
    __shared__ __bf16 sWt[KOUT * LDP];  // 34,816 B (KOUT=128)

    const int tid = threadIdx.x;
    const int rowBase = blockIdx.x * 64;

    // Stage W transposed: sWt[j*LDP + k] = W[k][j].  Coalesced global reads;
    // LDS write stride 272B => consecutive lanes 4 banks apart (no conflict).
    for (int i = tid; i < 128 * KOUT; i += 128) {
        int k = i / KOUT;
        int j = i % KOUT;
        sWt[j * LDP + k] = (__bf16)W[i];
    }

    // Stage A tile (64 x 128), coalesced, fused ns row-scale, zero-pad past N.
    for (int i = tid; i < 64 * 128; i += 128) {
        int r = i >> 7, c = i & 127;
        int gr = rowBase + r;
        float v = 0.0f;
        if (gr < N) v = H[(long long)gr * 128 + c] * ns[gr];
        sA[r * LDP + c] = (__bf16)v;
    }
    __syncthreads();

    const int wave = tid >> 5;       // 0..3 -> 16-row stripe
    const int lane = tid & 31;
    const int hb   = lane >> 4;      // wave half (ISA wave32 WMMA layout)
    const int l16  = lane & 15;

    constexpr int NCT = KOUT / 16;   // column tiles
    v8f acc[NCT];
    v8f zero = {};
#pragma unroll
    for (int ct = 0; ct < NCT; ++ct) acc[ct] = zero;

    const __bf16* aBase = &sA[(wave * 16 + l16) * LDP];

#pragma unroll
    for (int ks = 0; ks < 4; ++ks) {          // K = 128 in 4 x 32 steps
        // A operand (16-bit 16x32 layout): lane needs K = {hb*8+i} and
        // {16+hb*8+i}, i=0..7 -> two contiguous 16B runs -> 2x ds_load_b128.
        const v8bf* ap = (const v8bf*)(aBase + ks * 32 + hb * 8);
        v8bf a0 = ap[0];
        v8bf a1 = ap[2];                       // +16 bf16 elements
        v16bf a = __builtin_shufflevector(a0, a1, 0, 1, 2, 3, 4, 5, 6, 7,
                                          8, 9, 10, 11, 12, 13, 14, 15);

#pragma unroll
        for (int ct = 0; ct < NCT; ++ct) {
            // B operand (32x16): same K order as A: K = {hb*8+i} and
            //  {16+hb*8+i}, i=0..7, from sWt[(ct*16+l16)*LDP + ks*32 + K].
            const v8bf* bp =
                (const v8bf*)(&sWt[(ct * 16 + l16) * LDP + ks * 32 + hb * 8]);
            v8bf b0 = bp[0];
            v8bf b1 = bp[2];
            v16bf b = __builtin_shufflevector(b0, b1, 0, 1, 2, 3, 4, 5, 6, 7,
                                              8, 9, 10, 11, 12, 13, 14, 15);

            acc[ct] = __builtin_amdgcn_wmma_f32_16x16x32_bf16(
                false, a, false, b, (short)0, acc[ct], false, false);
        }
    }

    // D layout: VGPR r holds row (r + hb*8), column l16.
    const int mBase = rowBase + wave * 16 + hb * 8;
#pragma unroll
    for (int ct = 0; ct < NCT; ++ct) {
#pragma unroll
        for (int r = 0; r < 8; ++r) {
            int gr = mBase + r;
            if (gr < N) out[(long long)gr * KOUT + ct * 16 + l16] = acc[ct][r];
        }
    }
}

// ------------------------------ SpMM aggregate -----------------------------
// agg[dst[e]][:] += xw[src[e]][:]   one thread per (edge, 4-col chunk)
__global__ __launch_bounds__(256) void spmm_atomic_kernel(const float* __restrict__ xw,
                                                          const int* __restrict__ src,
                                                          const int* __restrict__ dst,
                                                          float* __restrict__ agg,
                                                          int E, int K) {
    int tid = blockIdx.x * blockDim.x + threadIdx.x;
    int tpe = K >> 2;                       // threads per edge
    if (tid >= E * tpe) return;
    int e = tid / tpe;
    int c = (tid - e * tpe) << 2;
    int s = src[e], d = dst[e];
    const float4 v = *(const float4*)(xw + (long long)s * K + c);
    float* p = agg + (long long)d * K + c;
    atomicAdd(p + 0, v.x);
    atomicAdd(p + 1, v.y);
    atomicAdd(p + 2, v.z);
    atomicAdd(p + 3, v.w);
}

// ------------------------------- finalize ----------------------------------
// out = (agg * nd[row]) + bias[col], optional relu
__global__ __launch_bounds__(256) void finalize_kernel(const float* __restrict__ agg,
                                                       const float* __restrict__ nd,
                                                       const float* __restrict__ bias,
                                                       float* __restrict__ out,
                                                       int N, int K, int relu) {
    int tid = blockIdx.x * blockDim.x + threadIdx.x;
    if (tid >= N * K) return;
    int n = tid / K;
    int c = tid - n * K;
    float v = agg[tid] * nd[n] + bias[c];
    if (relu) v = fmaxf(v, 0.0f);
    out[tid] = v;
}

// ------------------------------- launcher ----------------------------------

static inline unsigned cdiv(long long a, long long b) { return (unsigned)((a + b - 1) / b); }

extern "C" void kernel_launch(void* const* d_in, const int* in_sizes, int n_in,
                              void* d_out, int out_size, void* d_ws, size_t ws_size,
                              hipStream_t stream) {
    const float* feats = (const float*)d_in[0];
    const float* W0 = (const float*)d_in[1];
    const float* b0 = (const float*)d_in[2];
    const float* W1 = (const float*)d_in[3];
    const float* b1 = (const float*)d_in[4];
    const float* W2 = (const float*)d_in[5];
    const float* b2 = (const float*)d_in[6];
    const int*   src = (const int*)d_in[7];
    const int*   dst = (const int*)d_in[8];

    const int N = in_sizes[0] / 128;
    const int E = in_sizes[7];

    float* ws = (float*)d_ws;
    float* ns   = ws;                         // [N]  (src-side norm)
    float* nd   = ws + N;                     // [N]  (dst-side norm)
    float* bufA = ws + 2 * (size_t)N;         // [N x 128]
    float* bufB = bufA + (size_t)N * 128;     // [N x 128]
    float* bufC = bufB + (size_t)N * 128;     // [N x 128]

    const int T = 256;

    // Degrees -> symmetric norms (ns, nd stored contiguously; one norm pass).
    zero_f32_kernel<<<cdiv(2LL * N, T), T, 0, stream>>>(ns, 2LL * N);
    degree_kernel<<<cdiv(E, T), T, 0, stream>>>(src, dst, ns, nd, E);
    norm_kernel<<<cdiv(2LL * N, T), T, 0, stream>>>(ns, 2 * N);

    const unsigned gemmGrid = cdiv(N, 64);

    // ---- Layer 0: features -> bufA (relu) ----
    gemm_ns_kernel<128><<<gemmGrid, 128, 0, stream>>>(feats, W0, ns, bufB, N);
    zero_f32_kernel<<<cdiv((long long)N * 128, T), T, 0, stream>>>(bufC, (long long)N * 128);
    spmm_atomic_kernel<<<cdiv((long long)E * 32, T), T, 0, stream>>>(bufB, src, dst, bufC, E, 128);
    finalize_kernel<<<cdiv((long long)N * 128, T), T, 0, stream>>>(bufC, nd, b0, bufA, N, 128, 1);

    // ---- Layer 1: bufA -> bufA (relu) ----
    gemm_ns_kernel<128><<<gemmGrid, 128, 0, stream>>>(bufA, W1, ns, bufB, N);
    zero_f32_kernel<<<cdiv((long long)N * 128, T), T, 0, stream>>>(bufC, (long long)N * 128);
    spmm_atomic_kernel<<<cdiv((long long)E * 32, T), T, 0, stream>>>(bufB, src, dst, bufC, E, 128);
    finalize_kernel<<<cdiv((long long)N * 128, T), T, 0, stream>>>(bufC, nd, b1, bufA, N, 128, 1);

    // ---- Layer 2 (K=64): bufA -> d_out (no relu) ----
    gemm_ns_kernel<64><<<gemmGrid, 128, 0, stream>>>(bufA, W2, ns, bufB, N);
    zero_f32_kernel<<<cdiv((long long)N * 64, T), T, 0, stream>>>(bufC, (long long)N * 64);
    spmm_atomic_kernel<<<cdiv((long long)E * 16, T), T, 0, stream>>>(bufB, src, dst, bufC, E, 64);
    finalize_kernel<<<cdiv((long long)N * 64, T), T, 0, stream>>>(bufC, nd, b2, (float*)d_out, N, 64, 0);
}
